// GNNNodeClassifier_21363167330558
// MI455X (gfx1250) — matrix operand order repaired and checked by reading, not compiled, not run on hardware
//
#include <hip/hip_runtime.h>
#include <cstdint>

// ---------------------------------------------------------------------------
// GNN node classifier forward on gfx1250 (wave32, v_wmma_f32_16x16x32_f16).
// One wave owns a 16-row tile; every dense layer is [16,K]@[K,32] as K/32
// chained WMMAs with two f32 accumulators. BatchNorm affines are folded into
// the weights at launch (W' = diag(s)W as f16 in WMMA B-register layout,
// b' = b + tW), so the hot loops are: global_load_b128 row gather -> LDS ->
// v_cvt to f16 A tiles, v16h B loads, WMMA, exact-erf GELU, and native
// global_atomic_add_f32 scatter into the 12.8MB L2-resident accumulator.
// ---------------------------------------------------------------------------

typedef __attribute__((ext_vector_type(16))) _Float16 v16h;
typedef __attribute__((ext_vector_type(8)))  float    v8f;
typedef __attribute__((ext_vector_type(4)))  float    v4f;

#define GAS __attribute__((address_space(1)))

__device__ __forceinline__ float gld(const float* p)  { return *(const GAS float*)p; }
__device__ __forceinline__ v4f   gld4(const float* p) { return *(const GAS v4f*)p; }
__device__ __forceinline__ int   gldi(const int* p)   { return *(const GAS int*)p; }
__device__ __forceinline__ void  gst(float* p, float v)  { *(GAS float*)p = v; }
__device__ __forceinline__ void  gst4(float* p, v4f v)   { *(GAS v4f*)p = v; }
__device__ __forceinline__ void  gsth(_Float16* p, _Float16 v) { *(GAS _Float16*)p = v; }

// Native fp32 atomic add to global memory (no return -> STOREcnt path).
__device__ __forceinline__ void gatomic_add(float* p, float v) {
  asm volatile("global_atomic_add_f32 %0, %1, off" : : "v"(p), "v"(v) : "memory");
}

__device__ __forceinline__ void lds_fence() {
  asm volatile("s_wait_dscnt 0" ::: "memory");
}

__device__ __forceinline__ float gelu_exact(float v) {
  return 0.5f * v * (1.0f + erff(v * 0.70710678118654752f));
}

// A operand: 16x32 f16 tile from row-major LDS [16][LDK], K-chunk base kb.
// ISA layout (CDNA5 ISA §7.12.2):
// lane<16 : M=lane,    K = {kb+0..7,  kb+16..23}
// lane>=16: M=lane-16, K = {kb+8..15, kb+24..31}
template <int LDK>
__device__ __forceinline__ v16h make_A(const float* in, int kb) {
  const int lane = threadIdx.x & 31;
  const int hi = lane >> 4, m = lane & 15;
  v16h a;
#pragma unroll
  for (int e = 0; e < 16; ++e) {
    const int k = kb + (e & 7) + 8 * hi + 16 * (e >> 3);
    a[e] = (_Float16)in[m * LDK + k];
  }
  return a;
}

// B operand from pre-swizzled packed f16 weights: tile tIdx, lane-contiguous.
__device__ __forceinline__ v16h load_Bp(const _Float16* wp, int tIdx) {
  const int lane = threadIdx.x & 31;
  return *(const GAS v16h*)(wp + tIdx * 512 + lane * 16);
}

struct Pk { const _Float16 *w1, *w2; const float *b1, *b2; };

// out[16][32] (LDS) = gelu( in[16][DIN] @ Wp + bias )
template <int DIN>
__device__ __forceinline__ void dense32_gelu(const float* in, const _Float16* wp,
                                             const float* bias, float* out) {
  const int lane = threadIdx.x & 31;
  const int hi = lane >> 4, nlo = lane & 15;
  v8f c0 = {}; v8f c1 = {};
#pragma unroll
  for (int kb = 0; kb < DIN; kb += 32) {
    v16h a  = make_A<DIN>(in, kb);
    v16h b0 = load_Bp(wp, (kb >> 5) * 2 + 0);
    c0 = __builtin_amdgcn_wmma_f32_16x16x32_f16(false, a, false, b0,
                                                (short)0, c0, false, false);
    v16h b1 = load_Bp(wp, (kb >> 5) * 2 + 1);
    c1 = __builtin_amdgcn_wmma_f32_16x16x32_f16(false, a, false, b1,
                                                (short)0, c1, false, false);
  }
  // C/D layout: VGPR r -> row r + 8*hi, col = nlo (+ n0)
  const float bz0 = gld(bias + nlo), bz1 = gld(bias + 16 + nlo);
#pragma unroll
  for (int r = 0; r < 8; ++r) {
    out[(r + 8 * hi) * 32 + nlo]      = gelu_exact(c0[r] + bz0);
    out[(r + 8 * hi) * 32 + 16 + nlo] = gelu_exact(c1[r] + bz1);
  }
  lds_fence();
}

template <int DIN>
__device__ __forceinline__ void ffn_tile(const Pk& p, const float* in,
                                         float* tmp, float* out) {
  dense32_gelu<DIN>(in,  p.w1, p.b1, tmp);
  dense32_gelu<32>(tmp, p.w2, p.b2, out);
}

// ---------------------------------------------------------------------------
// Weight packing: W'[k][n] = s[k]*W[k][n] as f16 in WMMA B layout,
// b'[n] = b[n] + sum_k t[k]*W[k][n].  One block-row per matrix.
// ---------------------------------------------------------------------------

struct PackDesc {
  const float *w, *s, *t, *b;
  _Float16* dst; float* bdst;
  int din, ncol;
};
struct PackTable { PackDesc d[13]; };

__global__ void pack_kernel(PackTable tab) {
  const PackDesc d = tab.d[blockIdx.y];
  const int i = blockIdx.x * blockDim.x + threadIdx.x;
  if (i < d.din * d.ncol) {
    const int e = i & 15, lane = (i >> 4) & 31, tIdx = i >> 9;
    const int ntn = d.ncol >> 4;                 // col tiles
    const int kb = (tIdx / ntn) * 32, n0 = (tIdx % ntn) * 16;
    const int k = kb + (e & 7) + 8 * (lane >> 4) + 16 * (e >> 3);
    const int n = n0 + (lane & 15);
    float v = gld(d.w + k * d.ncol + n);
    if (d.s) v *= gld(d.s + k);
    gsth(d.dst + i, (_Float16)v);
  }
  if (i < d.ncol) {
    float acc = d.b ? gld(d.b + i) : 0.f;
    if (d.t)
      for (int k = 0; k < d.din; ++k)
        acc += gld(d.t + k) * gld(d.w + k * d.ncol + i);
    gst(d.bdst + i, acc);
  }
}

// ---------------------------------------------------------------------------
// Kernels (128 threads = 4 waves per block; one wave = one 16-row tile)
// ---------------------------------------------------------------------------

__global__ void preprocess_kernel(const float* __restrict__ nf, Pk p,
                                  float* __restrict__ x, int N) {
  __shared__ float smem[4 * (2048 + 512 + 512)];   // 48 KB
  const int wave = threadIdx.x >> 5, lane = threadIdx.x & 31;
  const int base = (blockIdx.x * 4 + wave) * 16;
  if (base >= N) return;
  float* in  = smem + wave * 2048;
  float* tmp = smem + 4 * 2048 + wave * 512;
  float* out = smem + 4 * 2048 + 4 * 512 + wave * 512;
  const int m = lane & 15, hi = lane >> 4;
  const int node = min(base + m, N - 1);
  const float* srow = nf + (size_t)node * 128 + hi * 64;
  float* drow = in + m * 128 + hi * 64;
#pragma unroll
  for (int i = 0; i < 16; ++i)
    *(v4f*)(drow + 4 * i) = gld4(srow + 4 * i);
  lds_fence();
  ffn_tile<128>(p, in, tmp, out);
  const int nd = base + m;
  if (nd < N) {
#pragma unroll
    for (int i = 0; i < 4; ++i)
      gst4(x + (size_t)nd * 32 + hi * 16 + 4 * i,
           *(const v4f*)(out + m * 32 + hi * 16 + 4 * i));
  }
}

__global__ void count_kernel(const int* __restrict__ tgt,
                             float* __restrict__ cnt, int E) {
  int i = blockIdx.x * blockDim.x + threadIdx.x;
  const int stride = gridDim.x * blockDim.x;
  for (; i < E; i += stride)
    gatomic_add(&cnt[gldi(tgt + i)], 1.0f);
}

// messages = ffn(x[src]) * ew ; atomic scatter-add into sums[tgt]
__global__ void edge_kernel(const float* __restrict__ x,
                            const int* __restrict__ tgt,
                            const int* __restrict__ src,
                            const float* __restrict__ ew, Pk p,
                            float* __restrict__ sums, int E) {
  __shared__ float smem[4 * 1536];                 // 24 KB
  const int wave = threadIdx.x >> 5, lane = threadIdx.x & 31;
  const int base = (blockIdx.x * 4 + wave) * 16;
  if (base >= E) return;
  float* in  = smem + wave * 1536;
  float* tmp = in + 512;
  float* out = in + 1024;
  const int m = lane & 15, hi = lane >> 4;
  const int e = min(base + m, E - 1);
  const int sn = gldi(src + e);
  const float* srow = x + (size_t)sn * 32 + hi * 16;
  float* drow = in + m * 32 + hi * 16;
#pragma unroll
  for (int i = 0; i < 4; ++i)
    *(v4f*)(drow + 4 * i) = gld4(srow + 4 * i);
  lds_fence();
  ffn_tile<32>(p, in, tmp, out);
  const int e2 = base + m;
  if (e2 < E) {
    const float w = gld(ew + e2);
    float* dst = sums + (size_t)gldi(tgt + e2) * 32 + hi * 16;
#pragma unroll
    for (int j = 0; j < 16; ++j)
      gatomic_add(dst + j, out[m * 32 + hi * 16 + j] * w);
  }
}

// h = l2norm(ffn(concat(x, sums/max(cnt,1)))) ; x = h + x  (in place)
__global__ void update_kernel(float* __restrict__ x,
                              const float* __restrict__ sums,
                              const float* __restrict__ cnt, Pk p, int N) {
  __shared__ float smem[4 * (1024 + 512 + 512)];   // 32 KB
  const int wave = threadIdx.x >> 5, lane = threadIdx.x & 31;
  const int base = (blockIdx.x * 4 + wave) * 16;
  if (base >= N) return;
  float* in  = smem + wave * 2048;
  float* tmp = in + 1024;
  float* out = in + 1536;
  const int m = lane & 15, hi = lane >> 4;
  const int node = min(base + m, N - 1);
#pragma unroll
  for (int i = 0; i < 4; ++i)
    *(v4f*)(in + m * 64 + hi * 16 + 4 * i) =
        gld4(x + (size_t)node * 32 + hi * 16 + 4 * i);
  const float rc = 1.0f / fmaxf(gld(cnt + node), 1.0f);
#pragma unroll
  for (int i = 0; i < 4; ++i) {
    v4f v = gld4(sums + (size_t)node * 32 + hi * 16 + 4 * i);
    float* ai = in + m * 64 + 32 + hi * 16 + 4 * i;
    ai[0] = v.x * rc; ai[1] = v.y * rc; ai[2] = v.z * rc; ai[3] = v.w * rc;
  }
  lds_fence();
  ffn_tile<64>(p, in, tmp, out);
  // row l2 norm: each half-wave lane holds 16 cols of row m; combine halves.
  float ss2 = 0.f;
#pragma unroll
  for (int j = 0; j < 16; ++j) {
    const float v = out[m * 32 + hi * 16 + j];
    ss2 += v * v;
  }
  ss2 += __shfl_xor(ss2, 16, 32);
  const float rn = rsqrtf(fmaxf(ss2, 1e-12f));
  const int nd = base + m;
  if (nd < N) {
#pragma unroll
    for (int j = 0; j < 16; ++j)
      gst(x + (size_t)nd * 32 + hi * 16 + j,
          out[m * 32 + hi * 16 + j] * rn + in[m * 64 + hi * 16 + j]);
  }
}

__global__ void post_kernel(const float* __restrict__ x, Pk p,
                            float* __restrict__ o, int N) {
  __shared__ float smem[4 * 1536];
  const int wave = threadIdx.x >> 5, lane = threadIdx.x & 31;
  const int base = (blockIdx.x * 4 + wave) * 16;
  if (base >= N) return;
  float* in  = smem + wave * 1536;
  float* tmp = in + 512;
  float* out = in + 1024;
  const int m = lane & 15, hi = lane >> 4;
  const int node = min(base + m, N - 1);
#pragma unroll
  for (int i = 0; i < 4; ++i)
    *(v4f*)(in + m * 32 + hi * 16 + 4 * i) =
        gld4(x + (size_t)node * 32 + hi * 16 + 4 * i);
  lds_fence();
  ffn_tile<32>(p, in, tmp, out);
  const int nd = base + m;
  if (nd < N) {
#pragma unroll
    for (int i = 0; i < 4; ++i)
      gst4(o + (size_t)nd * 32 + hi * 16 + 4 * i,
           *(const v4f*)(out + m * 32 + hi * 16 + 4 * i));
  }
}

// out[B,16] = emb[idx] @ w[32,16] + b   (single WMMA per 16-query tile)
__global__ void logits_kernel(const float* __restrict__ emb,
                              const int* __restrict__ idx,
                              const _Float16* __restrict__ wp,
                              const float* __restrict__ bias,
                              float* __restrict__ out, int B) {
  __shared__ float smem[4 * 512];
  const int wave = threadIdx.x >> 5, lane = threadIdx.x & 31;
  const int base = (blockIdx.x * 4 + wave) * 16;
  if (base >= B) return;
  float* in = smem + wave * 512;
  const int m = lane & 15, hi = lane >> 4;
  const int q = min(base + m, B - 1);
  const int node = gldi(idx + q);
#pragma unroll
  for (int i = 0; i < 4; ++i)
    *(v4f*)(in + m * 32 + hi * 16 + 4 * i) =
        gld4(emb + (size_t)node * 32 + hi * 16 + 4 * i);
  lds_fence();
  v16h a = make_A<32>(in, 0);
  v16h b = load_Bp(wp, 0);
  v8f c = {};
  c = __builtin_amdgcn_wmma_f32_16x16x32_f16(false, a, false, b,
                                             (short)0, c, false, false);
  const int n = lane & 15;
  const float bz = gld(bias + n);
#pragma unroll
  for (int r = 0; r < 8; ++r) {
    const int qq = base + r + 8 * hi;
    if (qq < B) gst(out + (size_t)qq * 16 + n, c[r] + bz);
  }
}

// ---------------------------------------------------------------------------
// Host side
// ---------------------------------------------------------------------------

struct FfnRaw { const float *s1,*t1,*w1,*b1,*s2,*t2,*w2,*b2; };

static inline FfnRaw mk_leaf(void* const* d_in, int b) {
  FfnRaw p;
  p.s1 = (const float*)d_in[b + 0]; p.t1 = (const float*)d_in[b + 1];
  p.w1 = (const float*)d_in[b + 2]; p.b1 = (const float*)d_in[b + 3];
  p.s2 = (const float*)d_in[b + 4]; p.t2 = (const float*)d_in[b + 5];
  p.w2 = (const float*)d_in[b + 6]; p.b2 = (const float*)d_in[b + 7];
  return p;
}

static inline FfnRaw mk_concat(const float* p, int din) {
  FfnRaw f;
  f.s1 = p;              f.t1 = p + din;
  f.w1 = p + 2 * din;    f.b1 = f.w1 + din * 32;
  f.s2 = f.b1 + 32;      f.t2 = f.s2 + 32;
  f.w2 = f.t2 + 32;      f.b2 = f.w2 + 32 * 32;
  return f;
}

extern "C" void kernel_launch(void* const* d_in, const int* in_sizes, int n_in,
                              void* d_out, int out_size, void* d_ws, size_t ws_size,
                              hipStream_t stream) {
  (void)out_size; (void)ws_size;
  const float* nf   = (const float*)d_in[0];
  const int*  edges = (const int*)d_in[1];
  const float* ew   = (const float*)d_in[2];
  const int*  qidx  = (const int*)d_in[3];

  FfnRaw pre, c1p, c1u, c2p, c2u, post;
  const float *lw, *lb;
  if (n_in >= 54) {           // pytree leaves flattened individually
    pre = mk_leaf(d_in, 4);  c1p = mk_leaf(d_in, 12); c1u = mk_leaf(d_in, 20);
    c2p = mk_leaf(d_in, 28); c2u = mk_leaf(d_in, 36); post = mk_leaf(d_in, 44);
    lw = (const float*)d_in[52]; lb = (const float*)d_in[53];
  } else {                    // each tuple passed as one concatenated block
    pre  = mk_concat((const float*)d_in[4], 128);
    c1p  = mk_concat((const float*)d_in[5], 32);
    c1u  = mk_concat((const float*)d_in[6], 64);
    c2p  = mk_concat((const float*)d_in[7], 32);
    c2u  = mk_concat((const float*)d_in[8], 64);
    post = mk_concat((const float*)d_in[9], 32);
    lw = (const float*)d_in[10]; lb = (const float*)d_in[11];
  }

  const int N = in_sizes[0] / 128;
  const int E = in_sizes[2];
  const int B = in_sizes[3];

  float* x    = (float*)d_ws;              // [N,32]
  float* sums = x + (size_t)N * 32;        // [N,32] (reused as post output)
  float* cnt  = sums + (size_t)N * 32;     // [N]

  // packed weights (f16, WMMA B layout) + folded biases (f32), 64B aligned
  const size_t packOff = (((size_t)N * 65 * 4) + 63) & ~(size_t)63;
  _Float16* pw = (_Float16*)((char*)d_ws + packOff);
  float* pb = (float*)(pw + 17920);
  // half offsets per matrix (each tile = 512 halfs)
  _Float16 *pw_pre1 = pw,          *pw_pre2 = pw + 4096;
  _Float16 *pw_1p1  = pw + 5120,   *pw_1p2  = pw + 6144;
  _Float16 *pw_1u1  = pw + 7168,   *pw_1u2  = pw + 9216;
  _Float16 *pw_2p1  = pw + 10240,  *pw_2p2  = pw + 11264;
  _Float16 *pw_2u1  = pw + 12288,  *pw_2u2  = pw + 14336;
  _Float16 *pw_po1  = pw + 15360,  *pw_po2  = pw + 16384;
  _Float16 *pw_log  = pw + 17408;

  PackTable tab;
  const FfnRaw* raw[6] = { &pre, &c1p, &c1u, &c2p, &c2u, &post };
  _Float16* wdst[12] = { pw_pre1, pw_pre2, pw_1p1, pw_1p2, pw_1u1, pw_1u2,
                         pw_2p1, pw_2p2, pw_2u1, pw_2u2, pw_po1, pw_po2 };
  const int dins[6] = { 128, 32, 64, 32, 64, 32 };
  for (int f = 0; f < 6; ++f) {
    tab.d[2*f]   = { raw[f]->w1, raw[f]->s1, raw[f]->t1, raw[f]->b1,
                     wdst[2*f],   pb + 64*f,      dins[f], 32 };
    tab.d[2*f+1] = { raw[f]->w2, raw[f]->s2, raw[f]->t2, raw[f]->b2,
                     wdst[2*f+1], pb + 64*f + 32, 32,      32 };
  }
  tab.d[12] = { lw, nullptr, nullptr, lb, pw_log, pb + 384, 32, 16 };

  Pk kpre = { pw_pre1, pw_pre2, pb + 0,   pb + 32  };
  Pk k1p  = { pw_1p1,  pw_1p2,  pb + 64,  pb + 96  };
  Pk k1u  = { pw_1u1,  pw_1u2,  pb + 128, pb + 160 };
  Pk k2p  = { pw_2p1,  pw_2p2,  pb + 192, pb + 224 };
  Pk k2u  = { pw_2u1,  pw_2u2,  pb + 256, pb + 288 };
  Pk kpo  = { pw_po1,  pw_po2,  pb + 320, pb + 352 };

  const int* tgt = edges;
  const int* src = edges + E;

  const int nblk = (N + 63) / 64;          // 4 tiles of 16 rows per block
  const int eblk = (E + 63) / 64;
  const int qblk = (B + 63) / 64;

  pack_kernel<<<dim3(16, 13), 256, 0, stream>>>(tab);
  hipMemsetAsync(cnt, 0, (size_t)N * sizeof(float), stream);
  count_kernel<<<512, 256, 0, stream>>>(tgt, cnt, E);
  preprocess_kernel<<<nblk, 128, 0, stream>>>(nf, kpre, x, N);

  // conv 1
  hipMemsetAsync(sums, 0, (size_t)N * 32 * sizeof(float), stream);
  edge_kernel<<<eblk, 128, 0, stream>>>(x, tgt, src, ew, k1p, sums, E);
  update_kernel<<<nblk, 128, 0, stream>>>(x, sums, cnt, k1u, N);

  // conv 2
  hipMemsetAsync(sums, 0, (size_t)N * 32 * sizeof(float), stream);
  edge_kernel<<<eblk, 128, 0, stream>>>(x, tgt, src, ew, k2p, sums, E);
  update_kernel<<<nblk, 128, 0, stream>>>(x, sums, cnt, k2u, N);

  // postprocess into sums buffer (no longer needed), then logits
  post_kernel<<<nblk, 128, 0, stream>>>(x, kpo, sums, N);
  logits_kernel<<<qblk, 128, 0, stream>>>(sums, qidx, pw_log, pb + 384,
                                          (float*)d_out, B);
}
